// AutoregTypeDecoder_72945724555618
// MI455X (gfx1250) — matrix operand order repaired and checked by reading, not compiled, not run on hardware
//
#include <hip/hip_runtime.h>
#include <hip/hip_bf16.h>
#include <math.h>

// Problem constants (from reference): T=64, B=128, H=256, C=256, L=3, K=8
#define TT 64
#define BB 128
#define HH 256
#define CCn 256
#define LLn 3
#define KK 8
#define BK (BB * KK)       // 1024 rows when beams folded into batch
#define BKH (BK * HH)      // 262144
#define NEGV (-1e30f)

typedef __attribute__((ext_vector_type(16))) __bf16 v16bf;
typedef __attribute__((ext_vector_type(8)))  __bf16 v8bf;
typedef __attribute__((ext_vector_type(8)))  float  v8f;

union V16U { v16bf v; v8bf h[2]; };

// ---------------------------------------------------------------------------
// WMMA fragment loaders (CDNA5 16-bit layouts, wave32)
// A: 16x32 (MxK). Lane L: m = L&15, half = L>>4.
//    elems 0..7  hold K = 8*half + 0..7
//    elems 8..15 hold K = 16 + 8*half + 0..7
// B: 32x16 (KxN), computing inp @ W^T with row-major W[out, in]:
//    B[k][n] = W[n_global*H + k]  -> lane n loads two runs of 8 bf16.
//    Lane L: n = L&15, half = L>>4 -> same K order as A
// ---------------------------------------------------------------------------
__device__ __forceinline__ v16bf load_a_frag(const __bf16* __restrict__ A,
                                             int row_base, int kb, int lane) {
  int m = lane & 15, hf = lane >> 4;
  const __bf16* p = A + (size_t)(row_base + m) * HH + kb + hf * 8;
  V16U u;
  u.h[0] = *(const v8bf*)(p);
  u.h[1] = *(const v8bf*)(p + 16);
  return u.v;
}

__device__ __forceinline__ v16bf load_b_frag(const __bf16* __restrict__ W,
                                             int col_base, int kb, int lane) {
  int n = lane & 15, hf = lane >> 4;
  const __bf16* p = W + (size_t)(col_base + n) * HH + kb + hf * 8;
  V16U u;
  u.h[0] = *(const v8bf*)(p);
  u.h[1] = *(const v8bf*)(p + 16);
  return u.v;
}

#define WMMA_BF16(ACC, A, B)                                                  \
  (ACC) = __builtin_amdgcn_wmma_f32_16x16x32_bf16(false, (A), false, (B),     \
                                                  (short)0, (ACC), false, false)

__device__ __forceinline__ float sigm(float x) { return 1.f / (1.f + __expf(-x)); }

// ---------------------------------------------------------------------------
// Fused LSTM layer: one wave (32 threads) per (32-row, 16-hidden-col) tile.
// Two 16-row M-tiles per wave so each gate-weight B fragment feeds TWO
// independent WMMAs (halves B traffic, breaks the load->wmma lockstep).
// All four gate tiles (i,f,g,o) share the A fragments; the full elementwise
// cell update runs in the epilogue. Optionally adds x[t_next] and emits the
// bf16 "state" used by the next step's projection GEMM.
// ---------------------------------------------------------------------------
__global__ __launch_bounds__(32)
void k_lstm_layer(const __bf16* __restrict__ inp,    // [M, H] bf16
                  const __bf16* __restrict__ hprev,  // [M, H] bf16
                  const __bf16* __restrict__ Wih,    // [4H, H] bf16 (layer l)
                  const __bf16* __restrict__ Whh,    // [4H, H] bf16 (layer l)
                  const float*  __restrict__ bih,    // [4H]
                  const float*  __restrict__ bhh,    // [4H]
                  const float*  __restrict__ cprev,  // [M, H] f32
                  float*        __restrict__ cnew,   // [M, H] f32
                  __bf16*       __restrict__ hnew,   // [M, H] bf16
                  __bf16*       state_out,           // [M, H] bf16 or null
                  const float*  x_add)               // [B, H] f32 or null
{
  const int lane = threadIdx.x;
  const int row_base = blockIdx.x * 32;
  const int col_base = blockIdx.y * 16;   // hidden-unit column tile

  v8f acc[2][4];
#pragma unroll
  for (int rt = 0; rt < 2; ++rt)
#pragma unroll
    for (int g = 0; g < 4; ++g)
#pragma unroll
      for (int r = 0; r < 8; ++r) acc[rt][g][r] = 0.f;

  // Combined K loop: g = inp @ Wih^T + hprev @ Whh^T (K = H = 256, 8 chunks)
#pragma unroll
  for (int kb = 0; kb < HH; kb += 32) {
    v16bf ax0 = load_a_frag(inp,   row_base,      kb, lane);
    v16bf ax1 = load_a_frag(inp,   row_base + 16, kb, lane);
    v16bf ah0 = load_a_frag(hprev, row_base,      kb, lane);
    v16bf ah1 = load_a_frag(hprev, row_base + 16, kb, lane);
#pragma unroll
    for (int g = 0; g < 4; ++g) {
      v16bf bx = load_b_frag(Wih, g * HH + col_base, kb, lane);
      WMMA_BF16(acc[0][g], ax0, bx);
      WMMA_BF16(acc[1][g], ax1, bx);
      v16bf bh = load_b_frag(Whh, g * HH + col_base, kb, lane);
      WMMA_BF16(acc[0][g], ah0, bh);
      WMMA_BF16(acc[1][g], ah1, bh);
    }
  }

  // Epilogue: gate order i,f,g,o (PyTorch). C/D layout: lane L holds
  // n = L&15, rows m = r + 8*(L>>4) in VGPR r.
  const int n  = lane & 15;
  const int mo = (lane >> 4) * 8;
  const int col = col_base + n;
  float bias[4];
#pragma unroll
  for (int g = 0; g < 4; ++g)
    bias[g] = bih[g * HH + col] + bhh[g * HH + col];

#pragma unroll
  for (int rt = 0; rt < 2; ++rt) {
#pragma unroll
    for (int r = 0; r < 8; ++r) {
      const int row = row_base + rt * 16 + mo + r;
      const size_t idx = (size_t)row * HH + col;
      float gi = acc[rt][0][r] + bias[0];
      float gf = acc[rt][1][r] + bias[1];
      float gg = acc[rt][2][r] + bias[2];
      float go = acc[rt][3][r] + bias[3];
      float cn = sigm(gf) * cprev[idx] + sigm(gi) * tanhf(gg);
      float hn = sigm(go) * tanhf(cn);
      cnew[idx] = cn;
      hnew[idx] = (__bf16)hn;
      if (state_out) {
        // rows are b*K + k; residual x[t_next][b, col]
        float xr = x_add[(size_t)(row >> 3) * HH + col];
        state_out[idx] = (__bf16)(hn + xr);
      }
    }
  }
}

// ---------------------------------------------------------------------------
// Projection GEMM: logits = state @ Wp^T + bp   ([1024,256] x [256,256])
// Same 2-row-tile widening: each Wp fragment feeds two WMMAs.
// ---------------------------------------------------------------------------
__global__ __launch_bounds__(32)
void k_proj(const __bf16* __restrict__ state,  // [BK, H] bf16
            const __bf16* __restrict__ Wpb,    // [C, H]  bf16
            const float*  __restrict__ bp,     // [C]
            float*        __restrict__ logits) // [BK, C] f32
{
  const int lane = threadIdx.x;
  const int rb = blockIdx.x * 32;
  const int cb = blockIdx.y * 16;
  v8f acc0, acc1;
#pragma unroll
  for (int r = 0; r < 8; ++r) { acc0[r] = 0.f; acc1[r] = 0.f; }
#pragma unroll
  for (int kb = 0; kb < HH; kb += 32) {
    v16bf a0 = load_a_frag(state, rb,      kb, lane);
    v16bf a1 = load_a_frag(state, rb + 16, kb, lane);
    v16bf b  = load_b_frag(Wpb, cb, kb, lane);
    WMMA_BF16(acc0, a0, b);
    WMMA_BF16(acc1, a1, b);
  }
  const int n = lane & 15;
  const int mo = (lane >> 4) * 8;
  const float bias = bp[cb + n];
#pragma unroll
  for (int r = 0; r < 8; ++r) {
    logits[(size_t)(rb + mo + r) * CCn + cb + n]      = acc0[r] + bias;
    logits[(size_t)(rb + 16 + mo + r) * CCn + cb + n] = acc1[r] + bias;
  }
}

// ---------------------------------------------------------------------------
// Per-sample: log-softmax per beam, joint = prefix + logp, top-8 of K*C=2048.
// One 256-thread block (8 wave32) per batch row; wave w owns beam w's LSE.
// ---------------------------------------------------------------------------
__global__ __launch_bounds__(256)
void k_topk(const float* __restrict__ logits,  // [BK, C]
            float* __restrict__ prefix,        // [BK]  (in/out)
            int*   __restrict__ prev,          // [BK]
            int*   __restrict__ cls)           // [BK]
{
  __shared__ float joint[KK * CCn];  // 2048
  __shared__ float pf[KK];
  __shared__ float rv[256];
  __shared__ int   ri[256];
  __shared__ float ov[KK];
  __shared__ int   oi[KK];

  const int b = blockIdx.x;
  const int tid = threadIdx.x;
  const int w = tid >> 5, lane = tid & 31;

  if (tid < KK) pf[tid] = prefix[b * KK + tid];
  __syncthreads();

  const float* lg = logits + (size_t)(b * KK + w) * CCn;
  float mx = -__builtin_inff();
  for (int c = lane; c < CCn; c += 32) mx = fmaxf(mx, lg[c]);
#pragma unroll
  for (int o = 16; o; o >>= 1) mx = fmaxf(mx, __shfl_xor(mx, o));
  float s = 0.f;
  for (int c = lane; c < CCn; c += 32) s += __expf(lg[c] - mx);
#pragma unroll
  for (int o = 16; o; o >>= 1) s += __shfl_xor(s, o);
  const float lse = mx + __logf(s);
  const float p = pf[w];
  for (int c = lane; c < CCn; c += 32) joint[w * CCn + c] = p + lg[c] - lse;
  __syncthreads();

  for (int i = 0; i < KK; ++i) {
    float bv = -__builtin_inff();
    int bi = 0x7fffffff;
    for (int j = tid; j < KK * CCn; j += 256) {
      float v = joint[j];
      if (v > bv) { bv = v; bi = j; }     // ascending j keeps lowest tie idx
    }
    rv[tid] = bv; ri[tid] = bi;
    __syncthreads();
    for (int st = 128; st; st >>= 1) {
      if (tid < st) {
        if (rv[tid + st] > rv[tid] ||
            (rv[tid + st] == rv[tid] && ri[tid + st] < ri[tid])) {
          rv[tid] = rv[tid + st]; ri[tid] = ri[tid + st];
        }
      }
      __syncthreads();
    }
    if (tid == 0) { ov[i] = rv[0]; oi[i] = ri[0]; joint[ri[0]] = -__builtin_inff(); }
    __syncthreads();
  }

  if (tid < KK) {
    prefix[b * KK + tid] = ov[tid];
    prev[b * KK + tid] = oi[tid] >> 8;    // idx / C
    cls[b * KK + tid]  = oi[tid] & 255;   // idx % C
  }
}

// ---------------------------------------------------------------------------
// Beam bookkeeping: gather h/c along surviving beams, build class embeddings,
// carry + append choices.  One thread per (b,k,h) element (262144 threads).
// ---------------------------------------------------------------------------
__global__ __launch_bounds__(256)
void k_gather(const __bf16* __restrict__ h, const float* __restrict__ c,
              __bf16* __restrict__ h2, float* __restrict__ c2,
              const __bf16* __restrict__ embb,
              const int* __restrict__ cls, const int* __restrict__ prev,
              const int* __restrict__ cho_src, int* __restrict__ cho_dst,
              __bf16* __restrict__ e, int t)
{
  const int idx = blockIdx.x * blockDim.x + threadIdx.x;
  if (idx >= BKH) return;
  const int hh = idx & (HH - 1);
  const int bk = idx >> 8;          // b*K + k
  const int b  = bk >> 3;
  const int pv = prev[bk];
  const size_t src = (size_t)(b * KK + pv) * HH + hh;
#pragma unroll
  for (int l = 0; l < LLn; ++l) {
    h2[(size_t)l * BKH + idx] = h[(size_t)l * BKH + src];
    c2[(size_t)l * BKH + idx] = c[(size_t)l * BKH + src];
  }
  e[idx] = embb[(size_t)cls[bk] * HH + hh];
  if (hh < TT) {
    int v = (hh == t) ? cls[bk] : cho_src[(b * KK + pv) * TT + hh];
    cho_dst[bk * TT + hh] = v;
  }
}

// ---------------------------------------------------------------------------
// Small utility kernels
// ---------------------------------------------------------------------------
__global__ void k_cvt(const float* __restrict__ s, __bf16* __restrict__ d, int n) {
  int i = blockIdx.x * blockDim.x + threadIdx.x;
  if (i < n) d[i] = (__bf16)s[i];
}
__global__ void k_zero_u32(unsigned int* __restrict__ d, int n) {
  int i = blockIdx.x * blockDim.x + threadIdx.x;
  if (i < n) d[i] = 0u;
}
__global__ void k_pinit(float* __restrict__ prefix) {
  int i = blockIdx.x * blockDim.x + threadIdx.x;
  if (i < BK) prefix[i] = ((i & (KK - 1)) == 0) ? 0.f : NEGV;
}
// Broadcast init state across beams: h[l,b,k]=h0[l,b]; c likewise; state=x[0]
__global__ __launch_bounds__(256)
void k_init_bcast(const __bf16* __restrict__ h0, const float* __restrict__ c0,
                  const __bf16* __restrict__ xb,
                  __bf16* __restrict__ h, float* __restrict__ c,
                  __bf16* __restrict__ state)
{
  const int idx = blockIdx.x * blockDim.x + threadIdx.x;
  if (idx >= BKH) return;
  const int hh = idx & (HH - 1);
  const int b  = (idx >> 8) >> 3;
  const size_t s0 = (size_t)b * HH + hh;
#pragma unroll
  for (int l = 0; l < LLn; ++l) {
    h[(size_t)l * BKH + idx] = h0[(size_t)l * BB * HH + s0];
    c[(size_t)l * BKH + idx] = c0[(size_t)l * BB * HH + s0];
  }
  state[idx] = xb[s0];   // x[0] broadcast
}
__global__ void k_final(const int* __restrict__ cho, const float* __restrict__ prefix,
                        float* __restrict__ out) {
  int i = blockIdx.x * blockDim.x + threadIdx.x;
  const int NC = BB * KK * TT;           // 65536 choices
  if (i < NC) out[i] = (float)cho[i];
  else if (i < NC + BK) out[i] = prefix[i - NC];
}

// ---------------------------------------------------------------------------
extern "C" void kernel_launch(void* const* d_in, const int* in_sizes, int n_in,
                              void* d_out, int out_size, void* d_ws, size_t ws_size,
                              hipStream_t stream) {
  (void)in_sizes; (void)n_in; (void)out_size; (void)ws_size;
  const float* x   = (const float*)d_in[0];   // [T,B,H]
  const float* emb = (const float*)d_in[1];   // [C,H]
  const float* Wih = (const float*)d_in[2];   // [L,4H,H]
  const float* Whh = (const float*)d_in[3];   // [L,4H,H]
  const float* bih = (const float*)d_in[4];   // [L,4H]
  const float* bhh = (const float*)d_in[5];   // [L,4H]
  const float* Wp  = (const float*)d_in[6];   // [C,H]
  const float* bp  = (const float*)d_in[7];   // [C]
  float* out = (float*)d_out;

  // Carve workspace
  uint8_t* base = (uint8_t*)d_ws;
  size_t off = 0;
  auto carve = [&](size_t bytes) -> void* {
    void* p = base + off;
    off += (bytes + 255) & ~(size_t)255;
    return p;
  };
  __bf16* xb    = (__bf16*)carve((size_t)TT * BB * HH * 2);
  __bf16* embb  = (__bf16*)carve((size_t)CCn * HH * 2);
  __bf16* Wihb  = (__bf16*)carve((size_t)LLn * 4 * HH * HH * 2);
  __bf16* Whhb  = (__bf16*)carve((size_t)LLn * 4 * HH * HH * 2);
  __bf16* Wpb   = (__bf16*)carve((size_t)CCn * HH * 2);
  __bf16* hbuf  = (__bf16*)carve((size_t)LLn * BKH * 2);
  __bf16* h2    = (__bf16*)carve((size_t)LLn * BKH * 2);
  float*  cbuf  = (float*)carve((size_t)LLn * BKH * 4);
  float*  c2    = (float*)carve((size_t)LLn * BKH * 4);
  __bf16* h0    = (__bf16*)carve((size_t)LLn * BB * HH * 2);
  float*  c0    = (float*)carve((size_t)LLn * BB * HH * 4);
  __bf16* hz    = (__bf16*)carve((size_t)BB * HH * 2);   // zeros
  float*  cz    = (float*)carve((size_t)BB * HH * 4);    // zeros
  __bf16* state = (__bf16*)carve((size_t)BKH * 2);
  __bf16* ebuf  = (__bf16*)carve((size_t)BKH * 2);
  float*  logits= (float*)carve((size_t)BK * CCn * 4);
  float*  prefix= (float*)carve((size_t)BK * 4);
  int*    prev  = (int*)carve((size_t)BK * 4);
  int*    cls   = (int*)carve((size_t)BK * 4);
  int*    cho0  = (int*)carve((size_t)BK * TT * 4);
  int*    cho1  = (int*)carve((size_t)BK * TT * 4);

  const int NB = 256;
  auto gr = [&](int n) { return dim3((n + NB - 1) / NB); };

  // --- one-time (per launch) conversions & init --------------------------
  k_cvt<<<gr(TT * BB * HH), NB, 0, stream>>>(x, xb, TT * BB * HH);
  k_cvt<<<gr(CCn * HH), NB, 0, stream>>>(emb, embb, CCn * HH);
  k_cvt<<<gr(LLn * 4 * HH * HH), NB, 0, stream>>>(Wih, Wihb, LLn * 4 * HH * HH);
  k_cvt<<<gr(LLn * 4 * HH * HH), NB, 0, stream>>>(Whh, Whhb, LLn * 4 * HH * HH);
  k_cvt<<<gr(CCn * HH), NB, 0, stream>>>(Wp, Wpb, CCn * HH);
  k_zero_u32<<<gr(BB * HH / 2), NB, 0, stream>>>((unsigned*)hz, BB * HH / 2);
  k_zero_u32<<<gr(BB * HH), NB, 0, stream>>>((unsigned*)cz, BB * HH);
  k_zero_u32<<<gr(BK * TT), NB, 0, stream>>>((unsigned*)cho0, BK * TT);
  k_zero_u32<<<gr(BK * TT), NB, 0, stream>>>((unsigned*)cho1, BK * TT);
  k_pinit<<<gr(BK), NB, 0, stream>>>(prefix);

  // get_rnn_init: one LSTM step on x[0] (M = B = 128) from zero state
  for (int l = 0; l < LLn; ++l) {
    const __bf16* inp = (l == 0) ? xb : (h0 + (size_t)(l - 1) * BB * HH);
    k_lstm_layer<<<dim3(BB / 32, HH / 16), 32, 0, stream>>>(
        inp, hz, Wihb + (size_t)l * 4 * HH * HH, Whhb + (size_t)l * 4 * HH * HH,
        bih + (size_t)l * 4 * HH, bhh + (size_t)l * 4 * HH,
        cz, c0 + (size_t)l * BB * HH, h0 + (size_t)l * BB * HH,
        nullptr, nullptr);
  }
  k_init_bcast<<<gr(BKH), NB, 0, stream>>>(h0, c0, xb, hbuf, cbuf, state);

  // --- 64 sequential beam-search steps -----------------------------------
  for (int t = 0; t < TT; ++t) {
    k_proj<<<dim3(BK / 32, CCn / 16), 32, 0, stream>>>(state, Wpb, bp, logits);
    k_topk<<<BB, 256, 0, stream>>>(logits, prefix, prev, cls);
    int* cs = (t & 1) ? cho1 : cho0;
    int* cd = (t & 1) ? cho0 : cho1;
    k_gather<<<gr(BKH), NB, 0, stream>>>(hbuf, cbuf, h2, c2, embb, cls, prev,
                                         cs, cd, ebuf, t);
    const int tn = (t + 1 < TT) ? (t + 1) : (TT - 1);
    for (int l = 0; l < LLn; ++l) {
      const __bf16* inp = (l == 0) ? ebuf : (hbuf + (size_t)(l - 1) * BKH);
      __bf16* so = (l == LLn - 1) ? state : nullptr;
      const float* xa = (l == LLn - 1) ? (x + (size_t)tn * BB * HH) : nullptr;
      k_lstm_layer<<<dim3(BK / 32, HH / 16), 32, 0, stream>>>(
          inp, h2 + (size_t)l * BKH,
          Wihb + (size_t)l * 4 * HH * HH, Whhb + (size_t)l * 4 * HH * HH,
          bih + (size_t)l * 4 * HH, bhh + (size_t)l * 4 * HH,
          c2 + (size_t)l * BKH, cbuf + (size_t)l * BKH, hbuf + (size_t)l * BKH,
          so, xa);
    }
  }

  // After t=63 the live choices buffer is cho0 (even # of ping-pongs).
  k_final<<<gr(BB * KK * TT + BK), NB, 0, stream>>>(cho0, prefix, out);
}
